// GCNConvLayer_20650202759169
// MI455X (gfx1250) — matrix operand order repaired and checked by reading, not compiled, not run on hardware
//
#include <hip/hip_runtime.h>
#include <hip/hip_bf16.h>

typedef __attribute__((ext_vector_type(16))) _Float16 v16h;
typedef __attribute__((ext_vector_type(8)))  float    v8f;

#define N_NODES  50000
#define N_EDGES  800000
#define D        128

// ---------------------------------------------------------------- utilities
__global__ void zero_f32(float* __restrict__ p, int n) {
    int i = blockIdx.x * blockDim.x + threadIdx.x;
    if (i < n) p[i] = 0.0f;
}

__global__ void degree_count(const int* __restrict__ dst, float* __restrict__ deg,
                             int n_edges) {
    int e = blockIdx.x * blockDim.x + threadIdx.x;
    if (e < n_edges) atomicAdd(&deg[dst[e]], 1.0f);
}

__global__ void node_norms(const float* __restrict__ deg,
                           float* __restrict__ node_norm,
                           float* __restrict__ inv_deg, int n) {
    int i = blockIdx.x * blockDim.x + threadIdx.x;
    if (i < n) {
        float d = deg[i] + 1.0f;        // in-degree + 1
        node_norm[i] = rsqrtf(d);       // d^-0.5
        inv_deg[i]   = 1.0f / d;        // for residual branch
    }
}

__global__ void convert_w_f16(const float* __restrict__ W,
                              _Float16* __restrict__ Wh, int n) {
    int i = blockIdx.x * blockDim.x + threadIdx.x;
    if (i < n) Wh[i] = (_Float16)W[i];
}

// ------------------------------------------------- h = nfeat @ W^T + b  (WMMA)
// One wave computes one 16x16 tile of h. 8 waves per block cover N = 128.
// Also writes the residual branch relu(h + res_w) * inv_deg into `out`,
// which the edge kernel then atomically accumulates into.
__global__ __launch_bounds__(256)
void gemm_h_residual(const float* __restrict__ nfeat,
                     const _Float16* __restrict__ Wh,      // W as f16, [D][D]
                     const float* __restrict__ bias,
                     const float* __restrict__ res_w,
                     const float* __restrict__ inv_deg,
                     float* __restrict__ h_out,
                     float* __restrict__ out) {
    const int lane = threadIdx.x & 31;
    const int wave = threadIdx.x >> 5;          // 0..7 : N tile index
    const int m0   = blockIdx.x * 16;           // 50000 = 3125 * 16 exactly
    const int n0   = wave * 16;
    const int m    = m0 + (lane & 15);
    const int n    = n0 + (lane & 15);
    const int hi   = lane >> 4;                 // lane half (ISA 7.12.2 layout)

    v8f c = {};

    const float*    arow = nfeat + (size_t)m * D;
    const _Float16* brow = Wh    + (size_t)n * D;   // row n of W == col n of W^T

    #pragma unroll
    for (int kt = 0; kt < 4; ++kt) {            // K = 128 = 4 * 32
        const int k0 = kt * 32;
        // A tile 16x32 f16: elements 0..15 -> K = k0+hi*16 .. +15
        v16h a;
        const float* ap = arow + k0 + (hi << 4);
        #pragma unroll
        for (int j = 0; j < 8; ++j) a[j]     = (_Float16)ap[j];
        #pragma unroll
        for (int j = 0; j < 8; ++j) a[8 + j] = (_Float16)ap[8 + j];

        // B tile 32x16 f16: B[k][n] = W[n][k] -> 16 contiguous halves of row n
        v16h bv = *(const v16h*)(brow + k0 + (hi << 4));

        // D = A x B + C : v_wmma_f32_16x16x32_f16
        c = __builtin_amdgcn_wmma_f32_16x16x32_f16(
                /*neg_a=*/false, a, /*neg_b=*/false, bv,
                /*c_mod=*/(short)0, c, /*reuse_a=*/false, /*reuse_b=*/false);
    }

    const float bn = bias[n];
    const float rw = res_w[n];
    #pragma unroll
    for (int r = 0; r < 8; ++r) {               // C/D layout: row = r + 8*hi
        const int   mr = m0 + r + (hi << 3);
        const float hv = c[r] + bn;
        h_out[(size_t)mr * D + n] = hv;
        out  [(size_t)mr * D + n] = fmaxf(hv + rw, 0.0f) * inv_deg[mr];
    }
}

// ------------------------------------------------- edge gather + atomic scatter
// 32 lanes per edge, 4 contiguous floats per lane (float4 gathers stay in L2:
// h and out are 25.6 MB each << 192 MB L2).
__global__ __launch_bounds__(256)
void edge_scatter(const float* __restrict__ h,
                  const int* __restrict__ src, const int* __restrict__ dst,
                  const int* __restrict__ efeat,
                  const float* __restrict__ node_norm,
                  const float* __restrict__ edge_embed,
                  float* __restrict__ out) {
    const long long idx = (long long)blockIdx.x * blockDim.x + threadIdx.x;
    const int e = (int)(idx >> 5);
    if (e >= N_EDGES) return;
    const int q  = ((int)idx & 31) << 2;        // feature offset (0,4,...,124)
    const int s  = src[e];
    const int t  = dst[e];
    const int et = efeat[e];
    const float nrm = node_norm[s] * node_norm[t];

    const float4 hv = *(const float4*)(h          + (size_t)s  * D + q);
    const float4 ev = *(const float4*)(edge_embed + (size_t)et * D + q);
    float* o = out + (size_t)t * D + q;
    atomicAdd(o + 0, nrm * fmaxf(hv.x + ev.x, 0.0f));
    atomicAdd(o + 1, nrm * fmaxf(hv.y + ev.y, 0.0f));
    atomicAdd(o + 2, nrm * fmaxf(hv.z + ev.z, 0.0f));
    atomicAdd(o + 3, nrm * fmaxf(hv.w + ev.w, 0.0f));
}

// --------------------------------------------------------------------- launch
extern "C" void kernel_launch(void* const* d_in, const int* in_sizes, int n_in,
                              void* d_out, int out_size, void* d_ws, size_t ws_size,
                              hipStream_t stream) {
    (void)in_sizes; (void)n_in; (void)out_size; (void)ws_size;
    const float* nfeat      = (const float*)d_in[0];
    const int*   efeat      = (const int*)  d_in[1];
    const int*   src        = (const int*)  d_in[2];
    const int*   dst        = (const int*)  d_in[3];
    const float* W          = (const float*)d_in[4];
    const float* bias       = (const float*)d_in[5];
    const float* edge_embed = (const float*)d_in[6];
    const float* res_w      = (const float*)d_in[7];
    float*       out        = (float*)d_out;

    // workspace layout (bytes)
    char* ws = (char*)d_ws;
    float*    h         = (float*)    ws;                  // 50000*128*4 = 25,600,000
    float*    deg       = (float*)   (ws + 25600000);      // 200,000
    float*    node_norm = (float*)   (ws + 25800000);      // 200,000
    float*    inv_deg   = (float*)   (ws + 26000000);      // 200,000
    _Float16* Wh        = (_Float16*)(ws + 26200000);      // 32,768

    zero_f32     <<<(N_NODES + 255) / 256, 256, 0, stream>>>(deg, N_NODES);
    degree_count <<<(N_EDGES + 255) / 256, 256, 0, stream>>>(dst, deg, N_EDGES);
    node_norms   <<<(N_NODES + 255) / 256, 256, 0, stream>>>(deg, node_norm, inv_deg, N_NODES);
    convert_w_f16<<<(D * D + 255)  / 256, 256, 0, stream>>>(W, Wh, D * D);

    gemm_h_residual<<<N_NODES / 16, 256, 0, stream>>>(nfeat, Wh, bias, res_w,
                                                      inv_deg, h, out);

    edge_scatter<<<(N_EDGES * 32) / 256, 256, 0, stream>>>(h, src, dst, efeat,
                                                           node_norm, edge_embed, out);
}
